// GPSModel_78718160601837
// MI455X (gfx1250) — matrix operand order repaired and checked by reading, not compiled, not run on hardware
//
#include <hip/hip_runtime.h>
#include <hip/hip_bf16.h>

typedef _Float16 half_t;
typedef __attribute__((ext_vector_type(16))) _Float16 v16h;
typedef __attribute__((ext_vector_type(8)))  _Float16 v8h;
typedef __attribute__((ext_vector_type(8)))  float    v8f;

#define NNODE 32768
#define NEDGE 524288
#define NGPH  64
#define NPGR  512
#define NLAY  3
#define NCH   64

// ---------------- workspace layout (bytes, 256-aligned) ----------------
#define OFF_H      0ull          // float  N*64      (8388608)
#define OFF_HF16   8388608ull    // half   N*64      (4194304)
#define OFF_AGGR   12582912ull   // float  N*64      (8388608)
#define OFF_A0     20971520ull   // half   N*64      (4194304)
#define OFF_HID    25165824ull   // half   N*128     (8388608)
#define OFF_H1     33554432ull   // float  N*64      (8388608)
#define OFF_QB     41943040ull   // half   N*64      (4194304)
#define OFF_KB     46137344ull   // half   N*64      (4194304)
#define OFF_VB     50331648ull   // half   N*64      (4194304)
#define OFF_OB     54525952ull   // half   N*64      (4194304)
#define OFF_OUT32  58720256ull   // float  N*64      (8388608)
#define OFF_OUTF16 67108864ull   // half   N*64      (4194304)
#define OFF_POOL   71303168ull   // float  64*64     (16384)
#define OFF_W      71319552ull   // half   weights   (294912)
#define W_PER_LAYER 49152        // halves per layer

// ---------------- WMMA helpers (wave32 layouts per CDNA5 ISA) ----------------
// A 16x32 f16: lane L (r=L&15, s=L>>4): e0..7 <-> K=s*8+e ; e8..15 <-> K=16+s*8+(e-8)
// B 32x16 f16: lane n (c=n&15, s=n>>4): e0..7 <-> K=s*8+e ; e8..15 <-> K=16+s*8+(e-8), column c
// C/D 16x16 f32: lane n (c=n&15, s=n>>4): element r <-> row M=r+8*s, col c

__device__ __forceinline__ v16h load_a_row(const half_t* __restrict__ Xrow, int ld, int lane) {
  const int r = lane & 15, s = lane >> 4;
  const half_t* p = Xrow + r * ld + s * 8;
  const v8h lo = *(const v8h*)p;
  const v8h hi = *(const v8h*)(p + 16);
  v16h a;
#pragma unroll
  for (int e = 0; e < 8; ++e) { a[e] = lo[e]; a[e + 8] = hi[e]; }
  return a;
}

__device__ __forceinline__ v16h load_b_wt(const half_t* __restrict__ Wt, int ld, int lane) {
  const int c = lane & 15, s = lane >> 4;
  const half_t* p = Wt + c * ld + s * 8;
  const v8h lo = *(const v8h*)p;
  const v8h hi = *(const v8h*)(p + 16);
  v16h b;
#pragma unroll
  for (int e = 0; e < 8; ++e) { b[e] = lo[e]; b[e + 8] = hi[e]; }
  return b;
}

template <int KD, int NT>
__device__ __forceinline__ void wave_gemm(const half_t* __restrict__ Xrow,
                                          const half_t* __restrict__ Wt,
                                          int lane, v8f (&acc)[NT]) {
#pragma unroll
  for (int kk = 0; kk < KD; kk += 32) {
    const v16h a = load_a_row(Xrow + kk, KD, lane);
#pragma unroll
    for (int ct = 0; ct < NT; ++ct) {
      const v16h b = load_b_wt(Wt + ct * 16 * KD + kk, KD, lane);
      acc[ct] = __builtin_amdgcn_wmma_f32_16x16x32_f16(false, a, false, b,
                                                       (short)0, acc[ct], false, false);
    }
  }
}

__device__ __forceinline__ float rowmax16(float v) {
  v = fmaxf(v, __shfl_xor(v, 1, 32));
  v = fmaxf(v, __shfl_xor(v, 2, 32));
  v = fmaxf(v, __shfl_xor(v, 4, 32));
  v = fmaxf(v, __shfl_xor(v, 8, 32));
  return v;
}
__device__ __forceinline__ float rowsum16(float v) {
  v += __shfl_xor(v, 1, 32);
  v += __shfl_xor(v, 2, 32);
  v += __shfl_xor(v, 4, 32);
  v += __shfl_xor(v, 8, 32);
  return v;
}

// ---------------- small utility kernels ----------------
__global__ void k_zero(float* p, int n) {
  for (int i = blockIdx.x * blockDim.x + threadIdx.x; i < n; i += gridDim.x * blockDim.x)
    p[i] = 0.f;
}

__global__ void k_wconv(const float* __restrict__ src, half_t* __restrict__ dst,
                        int O, int K, int transp) {
  int i = blockIdx.x * blockDim.x + threadIdx.x;
  if (i >= O * K) return;
  int o = i / K, k = i - o * K;
  float v = transp ? src[k * O + o] : src[o * K + k];
  dst[i] = (half_t)v;
}

__global__ void k_add_to_f16(const float* __restrict__ a, const float* __restrict__ b,
                             half_t* __restrict__ y, int n) {
  int i = blockIdx.x * blockDim.x + threadIdx.x;
  if (i < n) y[i] = (half_t)(a[i] + b[i]);
}

// ---------------- node embedding + PE projection ----------------
__global__ void k_embed(const int* __restrict__ x, const float* __restrict__ pe,
                        const float* __restrict__ node_emb,
                        const float* __restrict__ pe_g, const float* __restrict__ pe_b,
                        const float* __restrict__ plw, const float* __restrict__ plb,
                        float* __restrict__ h, half_t* __restrict__ hf16) {
  int n = blockIdx.x * blockDim.x + threadIdx.x;
  if (n >= NNODE) return;
  const float bns = rsqrtf(1.0f + 1e-5f);
  float pp[20];
#pragma unroll
  for (int k = 0; k < 20; ++k)
    pp[k] = pe[n * 20 + k] * (pe_g[k] * bns) + pe_b[k];
  int xi = x[n]; xi = xi < 0 ? 0 : (xi > 29 ? 29 : xi);
#pragma unroll
  for (int c = 0; c < 56; ++c) {
    float v = node_emb[xi * 56 + c];
    h[n * 64 + c] = v;
    hf16[n * 64 + c] = (half_t)v;
  }
#pragma unroll
  for (int j = 0; j < 8; ++j) {
    float a = plb[j];
#pragma unroll
    for (int k = 0; k < 20; ++k) a += pp[k] * plw[k * 8 + j];
    h[n * 64 + 56 + j] = a;
    hf16[n * 64 + 56 + j] = (half_t)a;
  }
}

// ---------------- edge message + segment-sum scatter (HW f32 atomics) ----------------
__global__ void k_edge(const int* __restrict__ ei, const int* __restrict__ eattr,
                       const float* __restrict__ edge_emb, const float* __restrict__ h,
                       float* __restrict__ aggr) {
  int tid = blockIdx.x * blockDim.x + threadIdx.x;
  int e = tid >> 4;
  if (e >= NEDGE) return;
  int c4 = (tid & 15) << 2;
  int src = ei[e];
  int dst = ei[NEDGE + e];
  int a = eattr[e]; a = a < 0 ? 0 : (a > 4 ? 4 : a);
  const float4 hv = *(const float4*)(h + (size_t)src * 64 + c4);
  const float4 ev = *(const float4*)(edge_emb + a * 64 + c4);
  float m0 = fmaxf(hv.x + ev.x, 0.f);
  float m1 = fmaxf(hv.y + ev.y, 0.f);
  float m2 = fmaxf(hv.z + ev.z, 0.f);
  float m3 = fmaxf(hv.w + ev.w, 0.f);
  float* d = aggr + (size_t)dst * 64 + c4;
  unsafeAtomicAdd(d + 0, m0);
  unsafeAtomicAdd(d + 1, m1);
  unsafeAtomicAdd(d + 2, m2);
  unsafeAtomicAdd(d + 3, m3);
}

// ---------------- GEMM kernels (one wave = 16 rows x full width) ----------------
// relu(X@Wt + b) -> f16, width NT*16  (GINE hidden & MLP hidden)
template <int KD, int NT>
__global__ __launch_bounds__(256) void k_gemm_relu(const half_t* __restrict__ X,
                                                   const half_t* __restrict__ Wt,
                                                   const float* __restrict__ bias,
                                                   half_t* __restrict__ Y) {
  const int wave = threadIdx.x >> 5, lane = threadIdx.x & 31;
  const int row0 = (blockIdx.x * 8 + wave) << 4;
  v8f acc[NT];
#pragma unroll
  for (int ct = 0; ct < NT; ++ct) acc[ct] = v8f{0.f,0.f,0.f,0.f,0.f,0.f,0.f,0.f};
  wave_gemm<KD, NT>(X + (size_t)row0 * KD, Wt, lane, acc);
  const int cn = lane & 15, rbase = row0 + ((lane >> 4) << 3);
#pragma unroll
  for (int ct = 0; ct < NT; ++ct) {
    const float bb = bias[ct * 16 + cn];
#pragma unroll
    for (int r = 0; r < 8; ++r) {
      float v = acc[ct][r] + bb;
      v = v > 0.f ? v : 0.f;
      Y[(size_t)(rbase + r) * (NT * 16) + ct * 16 + cn] = (half_t)v;
    }
  }
}

// z = hid@W2t + b2 ; h1 = bn1(z + h)
__global__ __launch_bounds__(256) void k_gemm_gine2(const half_t* __restrict__ X,
                                                    const half_t* __restrict__ Wt,
                                                    const float* __restrict__ bias,
                                                    const float* __restrict__ h,
                                                    const float* __restrict__ bg,
                                                    const float* __restrict__ bb,
                                                    float* __restrict__ h1) {
  const int wave = threadIdx.x >> 5, lane = threadIdx.x & 31;
  const int row0 = (blockIdx.x * 8 + wave) << 4;
  v8f acc[4];
#pragma unroll
  for (int ct = 0; ct < 4; ++ct) acc[ct] = v8f{0.f,0.f,0.f,0.f,0.f,0.f,0.f,0.f};
  wave_gemm<128, 4>(X + (size_t)row0 * 128, Wt, lane, acc);
  const int cn = lane & 15, rbase = row0 + ((lane >> 4) << 3);
  const float bns = rsqrtf(1.0f + 1e-5f);
#pragma unroll
  for (int ct = 0; ct < 4; ++ct) {
    const int c = ct * 16 + cn;
    const float bi = bias[c], sg = bg[c] * bns, sb = bb[c];
#pragma unroll
    for (int r = 0; r < 8; ++r) {
      const int row = rbase + r;
      float z = acc[ct][r] + bi + h[(size_t)row * 64 + c];
      h1[(size_t)row * 64 + c] = z * sg + sb;
    }
  }
}

// qkv = h@Wqkv^T + b ; scatter into [g][head][n][16] tiles, fold 1/sqrt(Dh) into Q
__global__ __launch_bounds__(256) void k_gemm_qkv(const half_t* __restrict__ X,
                                                  const half_t* __restrict__ Wt,
                                                  const float* __restrict__ bias,
                                                  half_t* __restrict__ Qb,
                                                  half_t* __restrict__ Kb,
                                                  half_t* __restrict__ Vb) {
  const int wave = threadIdx.x >> 5, lane = threadIdx.x & 31;
  const int row0 = (blockIdx.x * 8 + wave) << 4;
  v8f acc[12];
#pragma unroll
  for (int ct = 0; ct < 12; ++ct) acc[ct] = v8f{0.f,0.f,0.f,0.f,0.f,0.f,0.f,0.f};
  wave_gemm<64, 12>(X + (size_t)row0 * 64, Wt, lane, acc);
  const int cn = lane & 15, rbase = row0 + ((lane >> 4) << 3);
#pragma unroll
  for (int ct = 0; ct < 12; ++ct) {
    const int o = ct * 16 + cn;
    const float bi = bias[o];
    const int head = (o >> 4) & 3, d = o & 15, sec = o >> 6;
#pragma unroll
    for (int r = 0; r < 8; ++r) {
      const int row = rbase + r;
      const int g = row >> 9, nl = row & 511;
      const size_t idx = (size_t)(((g << 2) + head) << 9 | nl) * 16 + d;
      float v = acc[ct][r] + bi;
      if (sec == 0)      Qb[idx] = (half_t)(v * 0.25f);   // 1/sqrt(16)
      else if (sec == 1) Kb[idx] = (half_t)v;
      else               Vb[idx] = (half_t)v;
    }
  }
}

// o2 = Ob@Wout^T + b ; h2 = bn2(o2 + h) ; out = h1 + h2 (f32 + f16)
__global__ __launch_bounds__(256) void k_gemm_attnout(const half_t* __restrict__ X,
                                                      const half_t* __restrict__ Wt,
                                                      const float* __restrict__ bias,
                                                      const float* __restrict__ h,
                                                      const float* __restrict__ h1,
                                                      const float* __restrict__ bg,
                                                      const float* __restrict__ bb,
                                                      float* __restrict__ out32,
                                                      half_t* __restrict__ outf16) {
  const int wave = threadIdx.x >> 5, lane = threadIdx.x & 31;
  const int row0 = (blockIdx.x * 8 + wave) << 4;
  v8f acc[4];
#pragma unroll
  for (int ct = 0; ct < 4; ++ct) acc[ct] = v8f{0.f,0.f,0.f,0.f,0.f,0.f,0.f,0.f};
  wave_gemm<64, 4>(X + (size_t)row0 * 64, Wt, lane, acc);
  const int cn = lane & 15, rbase = row0 + ((lane >> 4) << 3);
  const float bns = rsqrtf(1.0f + 1e-5f);
#pragma unroll
  for (int ct = 0; ct < 4; ++ct) {
    const int c = ct * 16 + cn;
    const float bi = bias[c], sg = bg[c] * bns, sb = bb[c];
#pragma unroll
    for (int r = 0; r < 8; ++r) {
      const int row = rbase + r;
      float h2 = (acc[ct][r] + bi + h[(size_t)row * 64 + c]) * sg + sb;
      float o = h1[(size_t)row * 64 + c] + h2;
      out32[(size_t)row * 64 + c] = o;
      outf16[(size_t)row * 64 + c] = (half_t)o;
    }
  }
}

// y = m1@W2t + b ; h = bn3(out + y) -> f32 + f16
__global__ __launch_bounds__(256) void k_gemm_mlp2(const half_t* __restrict__ X,
                                                   const half_t* __restrict__ Wt,
                                                   const float* __restrict__ bias,
                                                   const float* __restrict__ out32,
                                                   const float* __restrict__ bg,
                                                   const float* __restrict__ bb,
                                                   float* __restrict__ h,
                                                   half_t* __restrict__ hf16) {
  const int wave = threadIdx.x >> 5, lane = threadIdx.x & 31;
  const int row0 = (blockIdx.x * 8 + wave) << 4;
  v8f acc[4];
#pragma unroll
  for (int ct = 0; ct < 4; ++ct) acc[ct] = v8f{0.f,0.f,0.f,0.f,0.f,0.f,0.f,0.f};
  wave_gemm<128, 4>(X + (size_t)row0 * 128, Wt, lane, acc);
  const int cn = lane & 15, rbase = row0 + ((lane >> 4) << 3);
  const float bns = rsqrtf(1.0f + 1e-5f);
#pragma unroll
  for (int ct = 0; ct < 4; ++ct) {
    const int c = ct * 16 + cn;
    const float bi = bias[c], sg = bg[c] * bns, sb = bb[c];
#pragma unroll
    for (int r = 0; r < 8; ++r) {
      const int row = rbase + r;
      float v = out32[(size_t)row * 64 + c] + acc[ct][r] + bi;
      v = v * sg + sb;
      h[(size_t)row * 64 + c] = v;
      hf16[(size_t)row * 64 + c] = (half_t)v;
    }
  }
}

// ---------------- flash-attention per (graph, head) ----------------
__global__ __launch_bounds__(128) void k_attn(const half_t* __restrict__ Qb,
                                              const half_t* __restrict__ Kb,
                                              const half_t* __restrict__ Vb,
                                              half_t* __restrict__ Ob) {
  __shared__ __align__(16) half_t Ks[512 * 16];   // K row-major [n][d]
  __shared__ __align__(16) half_t Vt[16 * 512];   // V transposed [d][n]
  __shared__ __align__(16) half_t Ps[4][16 * 32]; // per-wave P transpose scratch
  const int gh = blockIdx.x;
  const half_t* Qg = Qb + (size_t)gh * 512 * 16;
  const half_t* Kg = Kb + (size_t)gh * 512 * 16;
  const half_t* Vg = Vb + (size_t)gh * 512 * 16;
  const int t = threadIdx.x;
  {
    const float4* s4 = (const float4*)Kg;
    float4* d4 = (float4*)Ks;
    for (int i = t; i < 1024; i += 128) d4[i] = s4[i];
  }
  for (int i = t; i < 8192; i += 128) {
    int n = i >> 4, d = i & 15;
    Vt[d * 512 + n] = Vg[i];
  }
  __syncthreads();

  const int wave = t >> 5, lane = t & 31;
  const int cn = lane & 15, s = lane >> 4;
  const int g = gh >> 2, hh = gh & 3;
  half_t* sc = Ps[wave];

  for (int tile = wave; tile < 32; tile += 4) {
    const int row0 = tile << 4;
    v16h aq;
    {
      const v8h q = *(const v8h*)(Qg + (row0 + cn) * 16 + s * 8);
#pragma unroll
      for (int e = 0; e < 8; ++e) { aq[e] = q[e]; aq[e + 8] = (half_t)0.f; }
    }
    v8f o = v8f{0.f,0.f,0.f,0.f,0.f,0.f,0.f,0.f};
    float m[8], lsum[8];
#pragma unroll
    for (int r = 0; r < 8; ++r) { m[r] = -1e30f; lsum[r] = 0.f; }

    for (int ch = 0; ch < 512; ch += 32) {
      v16h b0, b1;
#pragma unroll
      for (int e = 0; e < 16; ++e) { b0[e] = (half_t)0.f; b1[e] = (half_t)0.f; }
      {  // e0..7 <-> d=s*8+e; e8..15 cover padded K-dim 16..31 -> zero
        const v8h k0a = *(const v8h*)(Ks + (ch + cn) * 16 + s * 8);
        const v8h k1a = *(const v8h*)(Ks + (ch + 16 + cn) * 16 + s * 8);
#pragma unroll
        for (int e = 0; e < 8; ++e) {
          b0[e] = k0a[e];
          b1[e] = k1a[e];
        }
      }
      const v8f zc = v8f{0.f,0.f,0.f,0.f,0.f,0.f,0.f,0.f};
      v8f s0 = __builtin_amdgcn_wmma_f32_16x16x32_f16(false, aq, false, b0, (short)0, zc, false, false);
      v8f s1 = __builtin_amdgcn_wmma_f32_16x16x32_f16(false, aq, false, b1, (short)0, zc, false, false);

#pragma unroll
      for (int r = 0; r < 8; ++r) {
        float rm = rowmax16(fmaxf(s0[r], s1[r]));
        float mn = fmaxf(m[r], rm);
        float alpha = __expf(m[r] - mn);
        float p0 = __expf(s0[r] - mn);
        float p1 = __expf(s1[r] - mn);
        float rs = rowsum16(p0 + p1);
        m[r] = mn;
        lsum[r] = lsum[r] * alpha + rs;
        o[r] = o[r] * alpha;
        const int M = r + (s << 3);
        sc[M * 32 + cn] = (half_t)p0;
        sc[M * 32 + 16 + cn] = (half_t)p1;
      }
      // reload P in A layout (same-wave LDS ops are in-order)
      v16h ap;
      {
        const v8h plo = *(const v8h*)(sc + cn * 32 + s * 8);
        const v8h phi = *(const v8h*)(sc + cn * 32 + 16 + s * 8);
#pragma unroll
        for (int e = 0; e < 8; ++e) { ap[e] = plo[e]; ap[e + 8] = phi[e]; }
      }
      v16h bv;
      {
        const v8h v0 = *(const v8h*)(Vt + cn * 512 + ch + s * 8);
        const v8h v1 = *(const v8h*)(Vt + cn * 512 + ch + 16 + s * 8);
#pragma unroll
        for (int e = 0; e < 8; ++e) { bv[e] = v0[e]; bv[e + 8] = v1[e]; }
      }
      o = __builtin_amdgcn_wmma_f32_16x16x32_f16(false, ap, false, bv, (short)0, o, false, false);
    }
#pragma unroll
    for (int r = 0; r < 8; ++r) {
      const int row = (g << 9) + row0 + r + (s << 3);
      Ob[(size_t)row * 64 + (hh << 4) + cn] = (half_t)(o[r] / lsum[r]);
    }
  }
}

// ---------------- pooling + readout ----------------
__global__ void k_pool(const float* __restrict__ h, float* __restrict__ pooled) {
  int g = blockIdx.x, c = threadIdx.x;
  float a = 0.f;
  for (int i = 0; i < NPGR; ++i)
    a += h[(size_t)((g << 9) + i) * 64 + c];
  pooled[g * 64 + c] = a;
}

__global__ void k_readout(const float* __restrict__ pooled,
                          const float* __restrict__ w1, const float* __restrict__ b1,
                          const float* __restrict__ w2, const float* __restrict__ b2,
                          const float* __restrict__ w3, const float* __restrict__ b3,
                          float* __restrict__ out) {
  int g = threadIdx.x;
  if (g >= NGPH) return;
  const float* p = pooled + g * 64;
  float r1[32];
#pragma unroll
  for (int j = 0; j < 32; ++j) {
    float a = b1[j];
    for (int k = 0; k < 64; ++k) a += p[k] * w1[k * 32 + j];
    r1[j] = fmaxf(a, 0.f);
  }
  float r2[16];
#pragma unroll
  for (int j = 0; j < 16; ++j) {
    float a = b2[j];
    for (int k = 0; k < 32; ++k) a += r1[k] * w2[k * 16 + j];
    r2[j] = fmaxf(a, 0.f);
  }
  float a = b3[0];
#pragma unroll
  for (int k = 0; k < 16; ++k) a += r2[k] * w3[k];
  out[g] = a;
}

// ---------------- host launch ----------------
extern "C" void kernel_launch(void* const* d_in, const int* in_sizes, int n_in,
                              void* d_out, int out_size, void* d_ws, size_t ws_size,
                              hipStream_t stream) {
  (void)in_sizes; (void)n_in; (void)out_size; (void)ws_size;
  char* ws = (char*)d_ws;
  float*  h      = (float*)(ws + OFF_H);
  half_t* hf16   = (half_t*)(ws + OFF_HF16);
  float*  aggr   = (float*)(ws + OFF_AGGR);
  half_t* a0     = (half_t*)(ws + OFF_A0);
  half_t* hid    = (half_t*)(ws + OFF_HID);
  float*  h1     = (float*)(ws + OFF_H1);
  half_t* Qb     = (half_t*)(ws + OFF_QB);
  half_t* Kb     = (half_t*)(ws + OFF_KB);
  half_t* Vb     = (half_t*)(ws + OFF_VB);
  half_t* Ob     = (half_t*)(ws + OFF_OB);
  float*  out32  = (float*)(ws + OFF_OUT32);
  half_t* outf16 = (half_t*)(ws + OFF_OUTF16);
  float*  pooled = (float*)(ws + OFF_POOL);
  half_t* wbuf   = (half_t*)(ws + OFF_W);

  const int*   x        = (const int*)d_in[0];
  const float* pe       = (const float*)d_in[1];
  const int*   ei       = (const int*)d_in[2];
  const int*   eattr    = (const int*)d_in[3];
  const float* node_emb = (const float*)d_in[5];
  const float* pe_g     = (const float*)d_in[6];
  const float* pe_b     = (const float*)d_in[7];
  const float* plw      = (const float*)d_in[8];
  const float* plb      = (const float*)d_in[9];
  const float* edge_emb = (const float*)d_in[10];

  // weight conversion: f16, B-operand layout [out][k]
  for (int l = 0; l < NLAY; ++l) {
    half_t* wl = wbuf + (size_t)l * W_PER_LAYER;
    k_wconv<<<(128*64+255)/256, 256, 0, stream>>>((const float*)d_in[11] + l*8192,  wl + 0,     128, 64, 1);
    k_wconv<<<(64*128+255)/256, 256, 0, stream>>>((const float*)d_in[13] + l*8192,  wl + 8192,  64, 128, 1);
    k_wconv<<<(192*64+255)/256, 256, 0, stream>>>((const float*)d_in[15] + l*12288, wl + 16384, 192, 64, 0);
    k_wconv<<<(64*64+255)/256,  256, 0, stream>>>((const float*)d_in[17] + l*4096,  wl + 28672, 64,  64, 0);
    k_wconv<<<(128*64+255)/256, 256, 0, stream>>>((const float*)d_in[25] + l*8192,  wl + 32768, 128, 64, 1);
    k_wconv<<<(64*128+255)/256, 256, 0, stream>>>((const float*)d_in[27] + l*8192,  wl + 40960, 64, 128, 1);
  }

  k_embed<<<NNODE/256, 256, 0, stream>>>(x, pe, node_emb, pe_g, pe_b, plw, plb, h, hf16);

  const int NV = NNODE * 64;
  for (int l = 0; l < NLAY; ++l) {
    half_t* wl = wbuf + (size_t)l * W_PER_LAYER;
    const float* gb1 = (const float*)d_in[12] + l*128;
    const float* gb2 = (const float*)d_in[14] + l*64;
    const float* aib = (const float*)d_in[16] + l*192;
    const float* aob = (const float*)d_in[18] + l*64;
    const float* b1g = (const float*)d_in[19] + l*64;
    const float* b1b = (const float*)d_in[20] + l*64;
    const float* b2g = (const float*)d_in[21] + l*64;
    const float* b2b = (const float*)d_in[22] + l*64;
    const float* b3g = (const float*)d_in[23] + l*64;
    const float* b3b = (const float*)d_in[24] + l*64;
    const float* mb1 = (const float*)d_in[26] + l*128;
    const float* mb2 = (const float*)d_in[28] + l*64;

    k_zero<<<2048, 256, 0, stream>>>(aggr, NV);
    k_edge<<<(NEDGE*16)/256, 256, 0, stream>>>(ei, eattr, edge_emb, h, aggr);
    k_add_to_f16<<<NV/256, 256, 0, stream>>>(h, aggr, a0, NV);

    k_gemm_relu<64, 8><<<256, 256, 0, stream>>>(a0, wl + 0, gb1, hid);
    k_gemm_gine2<<<256, 256, 0, stream>>>(hid, wl + 8192, gb2, h, b1g, b1b, h1);

    k_gemm_qkv<<<256, 256, 0, stream>>>(hf16, wl + 16384, aib, Qb, Kb, Vb);
    k_attn<<<NGPH * 4, 128, 0, stream>>>(Qb, Kb, Vb, Ob);
    k_gemm_attnout<<<256, 256, 0, stream>>>(Ob, wl + 28672, aob, h, h1, b2g, b2b, out32, outf16);

    k_gemm_relu<64, 8><<<256, 256, 0, stream>>>(outf16, wl + 32768, mb1, hid);
    k_gemm_mlp2<<<256, 256, 0, stream>>>(hid, wl + 40960, mb2, out32, b3g, b3b, h, hf16);
  }

  k_pool<<<NGPH, 64, 0, stream>>>(h, pooled);
  k_readout<<<1, 64, 0, stream>>>(pooled,
                                  (const float*)d_in[29], (const float*)d_in[30],
                                  (const float*)d_in[31], (const float*)d_in[32],
                                  (const float*)d_in[33], (const float*)d_in[34],
                                  (float*)d_out);
}
